// ModifiedGATPredictor_6236292513982
// MI455X (gfx1250) — matrix operand order repaired and checked by reading, not compiled, not run on hardware
//
#include <hip/hip_runtime.h>
#include <hip/hip_bf16.h>

typedef __attribute__((ext_vector_type(16))) _Float16 v16h;
typedef __attribute__((ext_vector_type(8)))  float    v8f;

#define D_IN 128
#define NH   4
#define NF   32

// ---------- order-preserving float<->uint encoding for atomic segment-max ----------
__device__ __forceinline__ unsigned enc_f32(float f) {
  unsigned u = __float_as_uint(f);
  return (u & 0x80000000u) ? ~u : (u | 0x80000000u);
}
__device__ __forceinline__ float dec_f32(unsigned k) {
  unsigned u = (k & 0x80000000u) ? (k & 0x7fffffffu) : ~k;
  return __uint_as_float(u);
}

// ---------- pack W [128,128] f32 row-major -> per-lane WMMA B layout, f16 ----------
// B operand (32x16 f16, wave32): lane n = L%16, group g = L/16,
// v16h element j (VGPR v=j/2, slot s=j%2) holds K = ks*32 + 8*g + j (j<8), ks*32 + 16 + 8*g + (j-8) (j>=8).
// Flat layout: Bp[((ct*4+ks)*32 + lane)*16 + j]
__global__ void pack_w_f16(const float* __restrict__ W, _Float16* __restrict__ Bp) {
  int idx = blockIdx.x * 256 + threadIdx.x;   // 0..1023
  if (idx >= 1024) return;
  int lane = idx & 31;
  int ks   = (idx >> 5) & 3;
  int ct   = idx >> 7;
  int g    = lane >> 4;
  int n    = ct * 16 + (lane & 15);
  _Float16* o = Bp + (size_t)idx * 16;
#pragma unroll
  for (int j = 0; j < 16; ++j) {
    int k = ks * 32 + (j < 8 ? 8 * g + j : 16 + 8 * g + (j - 8));
    o[j] = (_Float16)W[k * D_IN + n];
  }
}

// ---------- Z[N,128] = X[N,128] @ W[128,128] via v_wmma_f32_16x16x32_f16 ----------
// 8 waves / block; each wave computes a 16x128 output strip (8 col-tiles x 4 k-steps).
__global__ void __launch_bounds__(256)
gemm_wmma_f16(const float* __restrict__ X, const _Float16* __restrict__ Bp,
              float* __restrict__ Z, int nrows) {
  int wave  = threadIdx.x >> 5;
  int lane  = threadIdx.x & 31;
  int strip = blockIdx.x * 8 + wave;
  int nstrips = (nrows + 15) >> 4;
  if (strip >= nstrips) return;           // wave-uniform exit: EXEC stays all-1s for WMMA
  int m0 = strip << 4;
  bool full = (m0 + 16 <= nrows);         // wave-uniform: full strip -> unguarded stores
  int g  = lane >> 4;
  int mrow = m0 + (lane & 15);
  if (mrow >= nrows) mrow = nrows - 1;    // clamp loads (keeps full EXEC)
  const float* row = X + (size_t)mrow * D_IN;

  // A operand: 16x32 f16 per k-step. lane L: m = L%16; element j:
  //   j in 0..7  -> K = ks*32 + 8g + j
  //   j in 8..15 -> K = ks*32 + 16 + 8g + (j-8)
  v16h a[4];
#pragma unroll
  for (int ks = 0; ks < 4; ++ks) {
    const float4* p0 = (const float4*)(row + ks * 32 + g * 8);
    const float4* p1 = (const float4*)(row + ks * 32 + 16 + g * 8);
    float4 x0 = p0[0], x1 = p0[1];
    float4 y0 = p1[0], y1 = p1[1];
    a[ks][0]  = (_Float16)x0.x; a[ks][1]  = (_Float16)x0.y;
    a[ks][2]  = (_Float16)x0.z; a[ks][3]  = (_Float16)x0.w;
    a[ks][4]  = (_Float16)x1.x; a[ks][5]  = (_Float16)x1.y;
    a[ks][6]  = (_Float16)x1.z; a[ks][7]  = (_Float16)x1.w;
    a[ks][8]  = (_Float16)y0.x; a[ks][9]  = (_Float16)y0.y;
    a[ks][10] = (_Float16)y0.z; a[ks][11] = (_Float16)y0.w;
    a[ks][12] = (_Float16)y1.x; a[ks][13] = (_Float16)y1.y;
    a[ks][14] = (_Float16)y1.z; a[ks][15] = (_Float16)y1.w;
  }

#pragma unroll
  for (int ct = 0; ct < 8; ++ct) {
    v8f acc = {0.f, 0.f, 0.f, 0.f, 0.f, 0.f, 0.f, 0.f};
#pragma unroll
    for (int ks = 0; ks < 4; ++ks) {
      v16h b = *(const v16h*)(Bp + ((size_t)(ct * 4 + ks) * 32 + lane) * 16);
      acc = __builtin_amdgcn_wmma_f32_16x16x32_f16(false, a[ks], false, b,
                                                   (short)0, acc, false, false);
    }
    // C/D: lane L: n = ct*16 + L%16 ; VGPR r -> M = m0 + r + 8*(L/16)
    int ncol = ct * 16 + (lane & 15);
    float* zb = Z + (size_t)(m0 + 8 * g) * D_IN + ncol;
    if (full) {
#pragma unroll
      for (int r = 0; r < 8; ++r) zb[(size_t)r * D_IN] = acc[r];
    } else {
#pragma unroll
      for (int r = 0; r < 8; ++r) {
        if (m0 + 8 * g + r < nrows) zb[(size_t)r * D_IN] = acc[r];
      }
    }
  }
}

// ---------- el[n,h] = <z[n,h,:], al[h,:]> ; er likewise ----------
__global__ void attn_coeff(const float* __restrict__ Z, const float* __restrict__ al,
                           const float* __restrict__ ar, float* __restrict__ el,
                           float* __restrict__ er, int n_nodes) {
  int idx = blockIdx.x * blockDim.x + threadIdx.x;
  if (idx >= n_nodes * NH) return;
  int h = idx & 3;
  int n = idx >> 2;
  const float4* zp = (const float4*)(Z + (size_t)n * D_IN + h * NF);
  const float4* lp = (const float4*)(al + h * NF);
  const float4* rp = (const float4*)(ar + h * NF);
  float sl = 0.f, sr = 0.f;
#pragma unroll
  for (int i = 0; i < 8; ++i) {
    float4 z4 = zp[i], l4 = lp[i], r4 = rp[i];
    sl += z4.x * l4.x + z4.y * l4.y + z4.z * l4.z + z4.w * l4.w;
    sr += z4.x * r4.x + z4.y * r4.y + z4.z * r4.z + z4.w * r4.w;
  }
  el[idx] = sl;
  er[idx] = sr;
}

// ---------- zero/reset per-layer accumulators ----------
__global__ void init_layer(unsigned* __restrict__ m, float* __restrict__ s,
                           float* __restrict__ agg, int n_nodes) {
  int idx = blockIdx.x * blockDim.x + threadIdx.x;
  if (idx < n_nodes * NH) { m[idx] = 0u; s[idx] = 0.f; }   // 0 < enc(-inf): safe identity
  if (idx < n_nodes * D_IN) agg[idx] = 0.f;
}

// ---------- pass 1: leaky-relu logits + segment max (encoded atomicMax) ----------
__global__ void edge_logits(const int* __restrict__ src, const int* __restrict__ dst,
                            const float* __restrict__ el, const float* __restrict__ er,
                            float* __restrict__ eh, unsigned* __restrict__ m, int E) {
  int e = blockIdx.x * blockDim.x + threadIdx.x;
  if (e >= E) return;
  int sN = src[e], dN = dst[e];
  float4 l = *(const float4*)(el + (size_t)sN * NH);
  float4 r = *(const float4*)(er + (size_t)dN * NH);
  float4 v;
  v.x = l.x + r.x; v.y = l.y + r.y; v.z = l.z + r.z; v.w = l.w + r.w;
  v.x = v.x > 0.f ? v.x : 0.2f * v.x;
  v.y = v.y > 0.f ? v.y : 0.2f * v.y;
  v.z = v.z > 0.f ? v.z : 0.2f * v.z;
  v.w = v.w > 0.f ? v.w : 0.2f * v.w;
  *(float4*)(eh + (size_t)e * NH) = v;
  unsigned* mb = m + (size_t)dN * NH;
  atomicMax(mb + 0, enc_f32(v.x));
  atomicMax(mb + 1, enc_f32(v.y));
  atomicMax(mb + 2, enc_f32(v.z));
  atomicMax(mb + 3, enc_f32(v.w));
}

// ---------- pass 2: ex = exp(e - m[dst]) ; segment sum ----------
__global__ void edge_exp(const int* __restrict__ dst, float* __restrict__ eh,
                         const unsigned* __restrict__ m, float* __restrict__ s, int E) {
  int e = blockIdx.x * blockDim.x + threadIdx.x;
  if (e >= E) return;
  int dN = dst[e];
  float4 v = *(const float4*)(eh + (size_t)e * NH);
  uint4 mk = *(const uint4*)(m + (size_t)dN * NH);
  v.x = expf(v.x - dec_f32(mk.x));
  v.y = expf(v.y - dec_f32(mk.y));
  v.z = expf(v.z - dec_f32(mk.z));
  v.w = expf(v.w - dec_f32(mk.w));
  *(float4*)(eh + (size_t)e * NH) = v;
  float* sb = s + (size_t)dN * NH;
  atomicAdd(sb + 0, v.x);
  atomicAdd(sb + 1, v.y);
  atomicAdd(sb + 2, v.z);
  atomicAdd(sb + 3, v.w);
}

// ---------- pass 3: one wave per (edge, head); lane = feature ----------
// agg[dst,h,f] += attn * z[src,h,f]  -> warp-wide atomic covers 32 consecutive floats
__global__ void edge_aggregate(const int* __restrict__ src, const int* __restrict__ dst,
                               const float* __restrict__ eh, const float* __restrict__ s,
                               const float* __restrict__ Z, float* __restrict__ agg, int E) {
  int t = blockIdx.x * blockDim.x + threadIdx.x;
  int lane = t & 31;                 // feature f
  int p    = t >> 5;                 // (e,h) pair
  if (p >= E * NH) return;
  int h = p & 3;
  int e = p >> 2;
  int sN = src[e], dN = dst[e];      // same-address loads across the wave (broadcast)
  float attn = eh[p] / fmaxf(s[(size_t)dN * NH + h], 1e-9f);
  float val = attn * Z[(size_t)sN * D_IN + h * NF + lane];
  atomicAdd(agg + (size_t)dN * D_IN + h * NF + lane, val);
}

// ---------- layer-1 tail: residual + bias + ELU (flatten) ----------
__global__ void post_flatten_elu(const float* __restrict__ agg, const float* __restrict__ X,
                                 const float* __restrict__ b, float* __restrict__ out,
                                 int n_nodes) {
  int idx = blockIdx.x * blockDim.x + threadIdx.x;
  if (idx >= n_nodes * D_IN) return;
  int c = idx & (D_IN - 1);
  float v = agg[idx] + X[idx] + b[c];
  out[idx] = v > 0.f ? v : expm1f(v);
}

// ---------- layer-2 tail: residual + bias, mean over heads ----------
__global__ void post_mean(const float* __restrict__ agg, const float* __restrict__ X,
                          const float* __restrict__ b, float* __restrict__ out,
                          int n_nodes) {
  int idx = blockIdx.x * blockDim.x + threadIdx.x;
  if (idx >= n_nodes * NF) return;
  int f = idx & (NF - 1);
  int n = idx >> 5;
  float acc = 0.f;
#pragma unroll
  for (int h = 0; h < NH; ++h) {
    int c = h * NF + f;
    acc += agg[(size_t)n * D_IN + c] + X[(size_t)n * D_IN + c] + b[c];
  }
  out[idx] = acc * 0.25f;
}

// ---------- final projection [N,32] @ [32,1] + bp ----------
__global__ void proj_out(const float* __restrict__ h2, const float* __restrict__ Wp,
                         const float* __restrict__ bp, float* __restrict__ out, int n_nodes) {
  int n = blockIdx.x * blockDim.x + threadIdx.x;
  if (n >= n_nodes) return;
  const float4* hp = (const float4*)(h2 + (size_t)n * NF);
  const float4* wp = (const float4*)Wp;
  float acc = bp[0];
#pragma unroll
  for (int i = 0; i < 8; ++i) {
    float4 h4 = hp[i], w4 = wp[i];
    acc += h4.x * w4.x + h4.y * w4.y + h4.z * w4.z + h4.w * w4.w;
  }
  out[n] = acc;
}

extern "C" void kernel_launch(void* const* d_in, const int* in_sizes, int n_in,
                              void* d_out, int out_size, void* d_ws, size_t ws_size,
                              hipStream_t stream) {
  const float* feats = (const float*)d_in[0];
  const int*   src   = (const int*)d_in[1];
  const int*   dst   = (const int*)d_in[2];
  const float* W1    = (const float*)d_in[3];
  const float* al1   = (const float*)d_in[4];
  const float* ar1   = (const float*)d_in[5];
  const float* b1    = (const float*)d_in[6];
  const float* W2    = (const float*)d_in[7];
  const float* al2   = (const float*)d_in[8];
  const float* ar2   = (const float*)d_in[9];
  const float* b2    = (const float*)d_in[10];
  const float* Wp    = (const float*)d_in[11];
  const float* bp    = (const float*)d_in[12];

  const int N = out_size;        // 50000 rows, [N,1] output
  const int E = in_sizes[1];     // 800000 edges

  size_t off = 0;
  auto take = [&](size_t bytes) -> void* {
    void* p = (char*)d_ws + off;
    off += (bytes + 255) & ~(size_t)255;
    return p;
  };
  float*    z   = (float*)take((size_t)N * D_IN * sizeof(float));
  float*    h1  = (float*)take((size_t)N * D_IN * sizeof(float));
  float*    agg = (float*)take((size_t)N * D_IN * sizeof(float));
  float*    eh  = (float*)take((size_t)E * NH * sizeof(float));
  float*    el  = (float*)take((size_t)N * NH * sizeof(float));
  float*    er  = (float*)take((size_t)N * NH * sizeof(float));
  unsigned* m   = (unsigned*)take((size_t)N * NH * sizeof(unsigned));
  float*    s   = (float*)take((size_t)N * NH * sizeof(float));
  float*    h2  = (float*)take((size_t)N * NF * sizeof(float));
  _Float16* Bp  = (_Float16*)take((size_t)D_IN * D_IN * sizeof(_Float16));

  const int nstrips = (N + 15) / 16;
  dim3 blk(256);
  dim3 g_gemm((nstrips + 7) / 8);
  dim3 g_nh((N * NH + 255) / 256);
  dim3 g_nd((N * D_IN + 255) / 256);
  dim3 g_e((E + 255) / 256);
  dim3 g_agg((unsigned)(((size_t)E * NH * 32 + 255) / 256));
  dim3 g_nf((N * NF + 255) / 256);
  dim3 g_n((N + 255) / 256);

  // ---------------- Layer 1 ----------------
  pack_w_f16<<<4, blk, 0, stream>>>(W1, Bp);
  gemm_wmma_f16<<<g_gemm, blk, 0, stream>>>(feats, Bp, z, N);
  attn_coeff<<<g_nh, blk, 0, stream>>>(z, al1, ar1, el, er, N);
  init_layer<<<g_nd, blk, 0, stream>>>(m, s, agg, N);
  edge_logits<<<g_e, blk, 0, stream>>>(src, dst, el, er, eh, m, E);
  edge_exp<<<g_e, blk, 0, stream>>>(dst, eh, m, s, E);
  edge_aggregate<<<g_agg, blk, 0, stream>>>(src, dst, eh, s, z, agg, E);
  post_flatten_elu<<<g_nd, blk, 0, stream>>>(agg, feats, b1, h1, N);

  // ---------------- Layer 2 ----------------
  pack_w_f16<<<4, blk, 0, stream>>>(W2, Bp);
  gemm_wmma_f16<<<g_gemm, blk, 0, stream>>>(h1, Bp, z, N);
  attn_coeff<<<g_nh, blk, 0, stream>>>(z, al2, ar2, el, er, N);
  init_layer<<<g_nd, blk, 0, stream>>>(m, s, agg, N);
  edge_logits<<<g_e, blk, 0, stream>>>(src, dst, el, er, eh, m, E);
  edge_exp<<<g_e, blk, 0, stream>>>(dst, eh, m, s, E);
  edge_aggregate<<<g_agg, blk, 0, stream>>>(src, dst, eh, s, z, agg, E);
  post_mean<<<g_nf, blk, 0, stream>>>(agg, h1, b2, h2, N);

  // ---------------- Projection ----------------
  proj_out<<<g_n, blk, 0, stream>>>(h2, Wp, bp, (float*)d_out, N);
}
